// QwenMoe_56178172231929
// MI455X (gfx1250) — compile-verified, matrix operand order checked by reading, not run on hardware
//
#include <hip/hip_runtime.h>

// ---------------------------------------------------------------------------
// Qwen MoE block for MI455X (gfx1250, wave32, WMMA bf16 16x16x32)
// Sparse top-4 routing + shared expert. Memory-bound target ~65us
// (1.5 GB fp32 weights @ 23.3 TB/s); bf16 WMMA compute is far below roofline.
// ---------------------------------------------------------------------------

typedef __attribute__((ext_vector_type(4)))  __bf16 bf16x4;
typedef __attribute__((ext_vector_type(8)))  __bf16 bf16x8;
typedef __attribute__((ext_vector_type(16))) __bf16 bf16x16;
typedef __attribute__((ext_vector_type(8)))  float  f32x8;

constexpr int T   = 1024;   // tokens
constexpr int D   = 2048;   // hidden dim
constexpr int E   = 32;     // experts
constexpr int NI  = 1408;   // expert intermediate
constexpr int NIS = 5632;   // shared-expert intermediate
constexpr int MTL = 64;     // max 16-row token tiles per expert (T/16)

// workspace layout (bytes)
constexpr size_t OFF_CNT  = 0;                                   // 32 * i32
constexpr size_t OFF_POFF = 1024;                                // 32 * i32
constexpr size_t OFF_SEG  = 2048;                                // T * f32
constexpr size_t OFF_TOK  = 8192;                                // E*T * i32
constexpr size_t OFF_WL   = OFF_TOK + (size_t)E * T * 4;         // E*T * f32
constexpr size_t OFF_HB   = OFF_WL + (size_t)E * T * 4;          // T*D bf16
constexpr size_t OFF_HBUF = OFF_HB + (size_t)T * D * 2;          // 4608*NI bf16
constexpr size_t OFF_HSB  = OFF_HBUF + (size_t)4608 * NI * 2;    // T*NIS bf16

// -------------------- WMMA fragment helpers --------------------------------
// A layout (16-bit, 16x32): lanes 0-15 -> M=lane, K chunks {0..7, 16..23};
// lanes 16-31 -> M=lane-16, K chunks {8..15, 24..31}. B mirrors with lane=N.

__device__ __forceinline__ bf16x16 load_a_bf16(const __bf16* __restrict__ row,
                                               int k0, int half) {
  bf16x8 x = *(const bf16x8*)(row + k0 + half * 8);
  bf16x8 y = *(const bf16x8*)(row + k0 + half * 8 + 16);
  bf16x16 r;
#pragma unroll
  for (int i = 0; i < 8; ++i) { r[i] = x[i]; r[i + 8] = y[i]; }
  return r;
}

__device__ __forceinline__ bf16x16 load_b_f32(const float* __restrict__ row,
                                              int k0, int half) {
  f32x8 x = *(const f32x8*)(row + k0 + half * 8);
  f32x8 y = *(const f32x8*)(row + k0 + half * 8 + 16);
  bf16x16 r;
#pragma unroll
  for (int i = 0; i < 8; ++i) { r[i] = (__bf16)x[i]; r[i + 8] = (__bf16)y[i]; }
  return r;
}

__device__ __forceinline__ f32x8 wmma_bf16(bf16x16 a, bf16x16 b, f32x8 c) {
  return __builtin_amdgcn_wmma_f32_16x16x32_bf16(false, a, false, b,
                                                 (short)0, c, false, false);
}

// -------------------- small utility kernels --------------------------------

__global__ void zero_cnt_kernel(int* cnt) { cnt[threadIdx.x] = 0; }

__global__ void cvt_bf16_kernel(const float* __restrict__ x,
                                __bf16* __restrict__ y) {
  int i = (blockIdx.x * blockDim.x + threadIdx.x) * 4;
  float4 v = *(const float4*)(x + i);
  bf16x4 o;
  o[0] = (__bf16)v.x; o[1] = (__bf16)v.y; o[2] = (__bf16)v.z; o[3] = (__bf16)v.w;
  *(bf16x4*)(y + i) = o;
}

// Router: one wave32 per token, lane = expert. softmax over 32 logits,
// top-4 selection, scatter to per-expert lists. Also shared-expert sigmoid.
__global__ __launch_bounds__(256) void router_kernel(
    const float* __restrict__ H, const float* __restrict__ GW,
    const float* __restrict__ SEW, int* __restrict__ cnt,
    int* __restrict__ tok_list, float* __restrict__ w_list,
    float* __restrict__ seg) {
  const int lane = threadIdx.x & 31;
  const int t = blockIdx.x * 8 + (threadIdx.x >> 5);
  const float* h = H + (size_t)t * D;
  const float* g = GW + (size_t)lane * D;
  float acc = 0.f;
  for (int d = 0; d < D; d += 4) {
    float4 hv = *(const float4*)(h + d);
    float4 gv = *(const float4*)(g + d);
    acc = fmaf(hv.x, gv.x, acc); acc = fmaf(hv.y, gv.y, acc);
    acc = fmaf(hv.z, gv.z, acc); acc = fmaf(hv.w, gv.w, acc);
  }
  float mx = acc;
  for (int o = 16; o; o >>= 1) mx = fmaxf(mx, __shfl_xor(mx, o));
  float ex = __expf(acc - mx);
  float sm = ex;
  for (int o = 16; o; o >>= 1) sm += __shfl_xor(sm, o);
  float p = ex / sm;
#pragma unroll
  for (int k = 0; k < 4; ++k) {
    float bm = p; int bi = lane;
    for (int o = 16; o; o >>= 1) {
      float om = __shfl_xor(bm, o); int oi = __shfl_xor(bi, o);
      if (om > bm || (om == bm && oi < bi)) { bm = om; bi = oi; }
    }
    if (lane == 0) {
      int pos = atomicAdd(cnt + bi, 1);
      tok_list[bi * T + pos] = t;
      w_list[bi * T + pos] = bm;
    }
    if (lane == bi) p = -1.f;
  }
  // shared-expert sigmoid gate: sigmoid(h . se_gate_w)
  float sa = 0.f;
  for (int d = lane; d < D; d += 32) sa = fmaf(h[d], SEW[d], sa);
  for (int o = 16; o; o >>= 1) sa += __shfl_xor(sa, o);
  if (lane == 0) seg[t] = 1.f / (1.f + __expf(-sa));
}

// exclusive scan of 16-padded per-expert counts -> h-buffer row offsets
__global__ void scan_kernel(const int* __restrict__ cnt, int* __restrict__ poff) {
  int lane = threadIdx.x;
  int c = cnt[lane];
  int pad = (c + 15) & ~15;
  int x = pad;
  for (int o = 1; o < 32; o <<= 1) {
    int y = __shfl_up(x, o);
    if (lane >= o) x += y;
  }
  poff[lane] = x - pad;  // exclusive scan
}

// -------------------- expert gate/up + SiLU --------------------------------
// grid: (E*64, ceil(NI/256)); block 128 (4 waves, 16 tokens x 256 I cols).
// NI % 64 == 0, so each wave's 64-column slab is either fully valid or fully
// out of range -> single wave-level early exit keeps the K loop branch-free.
__global__ __launch_bounds__(128) void expert_up_kernel(
    const __bf16* __restrict__ Hb, const float* __restrict__ GPW,
    const float* __restrict__ UPW, const int* __restrict__ cnt,
    const int* __restrict__ poff, const int* __restrict__ tok_list,
    __bf16* __restrict__ hbuf) {
  const int e  = blockIdx.x >> 6;
  const int m0 = (blockIdx.x & 63) * 16;
  const int c  = cnt[e];
  if (m0 >= c) return;
  const int lane = threadIdx.x & 31;
  const int wid  = threadIdx.x >> 5;
  const int iBase = blockIdx.y * 256 + wid * 64;
  if (iBase + 64 > NI) return;   // wave-uniform exit (NI = 22*64)
  const int half = lane >> 4;
  const int l16  = lane & 15;
  const int m    = m0 + l16;
  const int tokv = tok_list[e * T + (m < c ? m : c - 1)];
  const __bf16* arow = Hb + (size_t)tokv * D;
  const float* gb = GPW + (size_t)e * NI * D;
  const float* ub = UPW + (size_t)e * NI * D;
  f32x8 ag[4] = {}; f32x8 au[4] = {};
  const float* growp[4]; const float* urowp[4];
#pragma unroll
  for (int n = 0; n < 4; ++n) {
    int ic = iBase + n * 16 + l16;
    growp[n] = gb + (size_t)ic * D;
    urowp[n] = ub + (size_t)ic * D;
  }
  for (int k0 = 0; k0 < D; k0 += 32) {
    bf16x16 a = load_a_bf16(arow, k0, half);
#pragma unroll
    for (int n = 0; n < 4; ++n) {
      bf16x16 bg = load_b_f32(growp[n], k0, half);
      ag[n] = wmma_bf16(a, bg, ag[n]);
      bf16x16 bu = load_b_f32(urowp[n], k0, half);
      au[n] = wmma_bf16(a, bu, au[n]);
    }
  }
  const int slotBase = poff[e] + m0;
#pragma unroll
  for (int n = 0; n < 4; ++n) {
    int i0 = iBase + n * 16;
#pragma unroll
    for (int r = 0; r < 8; ++r) {
      int M = r + half * 8;
      float gv = ag[n][r], uv = au[n][r];
      float hv = (gv / (1.f + __expf(-gv))) * uv;   // silu(g)*u
      hbuf[(size_t)(slotBase + M) * NI + i0 + l16] = (__bf16)hv;
    }
  }
}

// -------------------- shared-expert gate/up + SiLU -------------------------
// grid: (T/16, NIS/256); block 128
__global__ __launch_bounds__(128) void shared_up_kernel(
    const __bf16* __restrict__ Hb, const float* __restrict__ SGW,
    const float* __restrict__ SUW, __bf16* __restrict__ hsbuf) {
  const int t0   = blockIdx.x * 16;
  const int lane = threadIdx.x & 31;
  const int wid  = threadIdx.x >> 5;
  const int half = lane >> 4;
  const int l16  = lane & 15;
  const __bf16* arow = Hb + (size_t)(t0 + l16) * D;
  const int iBase = blockIdx.y * 256 + wid * 64;
  f32x8 ag[4] = {}; f32x8 au[4] = {};
  const float* growp[4]; const float* urowp[4];
#pragma unroll
  for (int n = 0; n < 4; ++n) {
    int ic = iBase + n * 16 + l16;
    growp[n] = SGW + (size_t)ic * D;
    urowp[n] = SUW + (size_t)ic * D;
  }
  for (int k0 = 0; k0 < D; k0 += 32) {
    bf16x16 a = load_a_bf16(arow, k0, half);
#pragma unroll
    for (int n = 0; n < 4; ++n) {
      bf16x16 bg = load_b_f32(growp[n], k0, half);
      ag[n] = wmma_bf16(a, bg, ag[n]);
      bf16x16 bu = load_b_f32(urowp[n], k0, half);
      au[n] = wmma_bf16(a, bu, au[n]);
    }
  }
#pragma unroll
  for (int n = 0; n < 4; ++n) {
    int i0 = iBase + n * 16;
#pragma unroll
    for (int r = 0; r < 8; ++r) {
      int M = r + half * 8;
      float gv = ag[n][r], uv = au[n][r];
      float hv = (gv / (1.f + __expf(-gv))) * uv;
      hsbuf[(size_t)(t0 + M) * NIS + i0 + l16] = (__bf16)hv;
    }
  }
}

// -------------------- shared-expert down-proj (initializes out) ------------
// grid: (T/16, D/256); block 128
__global__ __launch_bounds__(128) void shared_down_kernel(
    const __bf16* __restrict__ hsbuf, const float* __restrict__ SDW,
    const float* __restrict__ seg, float* __restrict__ out) {
  const int t0 = blockIdx.x * 16;
  __shared__ float s_seg[16];
  if (threadIdx.x < 16) s_seg[threadIdx.x] = seg[t0 + threadIdx.x];
  __syncthreads();
  const int lane = threadIdx.x & 31;
  const int wid  = threadIdx.x >> 5;
  const int half = lane >> 4;
  const int l16  = lane & 15;
  const __bf16* arow = hsbuf + (size_t)(t0 + l16) * NIS;
  const int dBase = blockIdx.y * 256 + wid * 64;
  f32x8 acc[4] = {};
  const float* brow[4];
#pragma unroll
  for (int n = 0; n < 4; ++n)
    brow[n] = SDW + (size_t)(dBase + n * 16 + l16) * NIS;
  for (int k0 = 0; k0 < NIS; k0 += 32) {
    bf16x16 a = load_a_bf16(arow, k0, half);
#pragma unroll
    for (int n = 0; n < 4; ++n) {
      bf16x16 b = load_b_f32(brow[n], k0, half);
      acc[n] = wmma_bf16(a, b, acc[n]);
    }
  }
#pragma unroll
  for (int n = 0; n < 4; ++n) {
#pragma unroll
    for (int r = 0; r < 8; ++r) {
      int M = r + half * 8;
      out[(size_t)(t0 + M) * D + dBase + n * 16 + l16] = s_seg[M] * acc[n][r];
    }
  }
}

// -------------------- expert down-proj + weighted combine ------------------
// grid: (E*64, D/256); block 128; atomically adds into out.
__global__ __launch_bounds__(128) void expert_down_kernel(
    const __bf16* __restrict__ hbuf, const float* __restrict__ DPW,
    const int* __restrict__ cnt, const int* __restrict__ poff,
    const int* __restrict__ tok_list, const float* __restrict__ w_list,
    float* __restrict__ out) {
  const int e  = blockIdx.x >> 6;
  const int m0 = (blockIdx.x & 63) * 16;
  const int c  = cnt[e];
  if (m0 >= c) return;
  __shared__ int   s_tok[16];
  __shared__ float s_w[16];
  if (threadIdx.x < 16) {
    int m = m0 + threadIdx.x;
    if (m < c) { s_tok[threadIdx.x] = tok_list[e * T + m];
                 s_w[threadIdx.x]   = w_list[e * T + m]; }
    else       { s_tok[threadIdx.x] = 0; s_w[threadIdx.x] = 0.f; }
  }
  __syncthreads();
  const int lane = threadIdx.x & 31;
  const int wid  = threadIdx.x >> 5;
  const int half = lane >> 4;
  const int l16  = lane & 15;
  const int slotBase = poff[e] + m0;
  const __bf16* arow = hbuf + (size_t)(slotBase + l16) * NI;
  const int dBase = blockIdx.y * 256 + wid * 64;
  const float* db = DPW + (size_t)e * D * NI;
  f32x8 acc[4] = {};
  const float* brow[4];
#pragma unroll
  for (int n = 0; n < 4; ++n)
    brow[n] = db + (size_t)(dBase + n * 16 + l16) * NI;
  for (int k0 = 0; k0 < NI; k0 += 32) {
    bf16x16 a = load_a_bf16(arow, k0, half);
#pragma unroll
    for (int n = 0; n < 4; ++n) {
      bf16x16 b = load_b_f32(brow[n], k0, half);
      acc[n] = wmma_bf16(a, b, acc[n]);
    }
  }
#pragma unroll
  for (int n = 0; n < 4; ++n) {
#pragma unroll
    for (int r = 0; r < 8; ++r) {
      int M = r + half * 8;
      float w = s_w[M];
      if (w != 0.f)
        atomicAdd(&out[(size_t)s_tok[M] * D + dBase + n * 16 + l16],
                  w * acc[n][r]);
    }
  }
}

// ---------------------------------------------------------------------------

extern "C" void kernel_launch(void* const* d_in, const int* in_sizes, int n_in,
                              void* d_out, int out_size, void* d_ws, size_t ws_size,
                              hipStream_t stream) {
  const float* H   = (const float*)d_in[0];   // [T, D]
  const float* GW  = (const float*)d_in[1];   // [E, D]
  const float* GPW = (const float*)d_in[2];   // [E, NI, D]
  const float* UPW = (const float*)d_in[3];   // [E, NI, D]
  const float* DPW = (const float*)d_in[4];   // [E, D, NI]
  const float* SGW = (const float*)d_in[5];   // [NIS, D]
  const float* SUW = (const float*)d_in[6];   // [NIS, D]
  const float* SDW = (const float*)d_in[7];   // [D, NIS]
  const float* SEW = (const float*)d_in[8];   // [1, D]
  float* out = (float*)d_out;                 // [T, D]

  uint8_t* ws = (uint8_t*)d_ws;
  int*    cnt      = (int*)(ws + OFF_CNT);
  int*    poff     = (int*)(ws + OFF_POFF);
  float*  seg      = (float*)(ws + OFF_SEG);
  int*    tok_list = (int*)(ws + OFF_TOK);
  float*  w_list   = (float*)(ws + OFF_WL);
  __bf16* Hb       = (__bf16*)(ws + OFF_HB);
  __bf16* hbuf     = (__bf16*)(ws + OFF_HBUF);
  __bf16* hsbuf    = (__bf16*)(ws + OFF_HSB);

  zero_cnt_kernel<<<1, 32, 0, stream>>>(cnt);
  cvt_bf16_kernel<<<(T * D / 4) / 256, 256, 0, stream>>>(H, Hb);
  router_kernel<<<T / 8, 256, 0, stream>>>(H, GW, SEW, cnt, tok_list, w_list, seg);
  scan_kernel<<<1, 32, 0, stream>>>(cnt, poff);
  expert_up_kernel<<<dim3(E * MTL, (NI + 255) / 256), 128, 0, stream>>>(
      Hb, GPW, UPW, cnt, poff, tok_list, hbuf);
  shared_up_kernel<<<dim3(T / 16, NIS / 256), 128, 0, stream>>>(Hb, SGW, SUW, hsbuf);
  shared_down_kernel<<<dim3(T / 16, D / 256), 128, 0, stream>>>(hsbuf, SDW, seg, out);
  expert_down_kernel<<<dim3(E * MTL, D / 256), 128, 0, stream>>>(
      hbuf, DPW, cnt, poff, tok_list, w_list, out);
}
